// SelfAttention_50706383897105
// MI455X (gfx1250) — matrix operand order repaired and checked by reading, not compiled, not run on hardware
//
#include <hip/hip_runtime.h>
#include <hip/hip_bf16.h>

// Self-attention for MI455X (gfx1250), wave32 + WMMA bf16 + TDM staging.
//   B=4, S=2048, D=1024, H=16, d_head=64, causal_mask=0.
// Pipeline: [gemm qkv (bf16 out)] -> [flash attention (bf16 out)] -> [gemm out (f32)]
// Matrix math via V_WMMA_F32_16X16X32_BF16; Q/K attention tiles staged with
// TENSOR_LOAD_TO_LDS (Tensor Data Mover) + s_wait_tensorcnt.

typedef __attribute__((ext_vector_type(16))) __bf16 v16bf;
typedef __attribute__((ext_vector_type(8)))  float  v8f;
typedef __attribute__((ext_vector_type(4)))  unsigned v4u;
typedef __attribute__((ext_vector_type(8)))  int    v8i;
typedef __attribute__((ext_vector_type(4)))  int    v4i;

#define DEV __device__ __forceinline__

DEV unsigned short f2bf(float f) {
  union { __bf16 h; unsigned short u; } c;
  c.h = (__bf16)f;                                  // hardware cvt, RTNE
  return c.u;
}

DEV unsigned pack2bf(float a, float b) {            // -> v_cvt_pk_bf16_f32
  union { __bf16 h[2]; unsigned u; } c;
  c.h[0] = (__bf16)a; c.h[1] = (__bf16)b;
  return c.u;
}

// A-fragment (16x32 bf16, M x K): lane holds row M = m + (lane&15).
// Lanes 0-15:  elems 0..7 -> K 0..7,  elems 8..15 -> K 16..23
// Lanes 16-31: elems 0..7 -> K 8..15, elems 8..15 -> K 24..31
DEV v16bf load_a_frag(const unsigned short* lds, int stride, int m, int kbase, int lane) {
  const unsigned short* p = lds + (m + (lane & 15)) * stride + kbase + ((lane & 16) ? 8 : 0);
  union { v16bf v; unsigned short u[16]; } r;
#pragma unroll
  for (int i = 0; i < 8; ++i) r.u[i] = p[i];
#pragma unroll
  for (int i = 0; i < 8; ++i) r.u[8 + i] = p[16 + i];
  return r.v;
}

// B-fragment (32x16 bf16, K x N) from an LDS tile stored N-major (tile[n][k]).
// lane holds column N = n + (lane&15); lanes 0-15: K 0..7,16..23, lanes 16-31: K 8..15,24..31.
DEV v16bf load_b_frag(const unsigned short* lds, int stride, int n, int kbase, int lane) {
  const unsigned short* p = lds + (n + (lane & 15)) * stride + kbase + ((lane & 16) ? 8 : 0);
  union { v16bf v; unsigned short u[16]; } r;
#pragma unroll
  for (int i = 0; i < 8; ++i) r.u[i] = p[i];
#pragma unroll
  for (int i = 0; i < 8; ++i) r.u[8 + i] = p[16 + i];
  return r.v;
}

DEV v8f wmma_bf16(v16bf a, v16bf b, v8f c) {
  return __builtin_amdgcn_wmma_f32_16x16x32_bf16(false, a, false, b, (short)0, c, false, false);
}

// ---------------------------------------------------------------------------
// Tensor Data Mover: 2D tile (tile_w elems of 2 bytes) x (tile_h rows), row
// stride stride_elems, global -> LDS at lds_addr.  Issue from one wave only;
// TDM ignores EXEC and tracks completion on the issuing wave's TENSORcnt.
// ---------------------------------------------------------------------------
DEV void tdm_wait0() {
#if __has_builtin(__builtin_amdgcn_s_wait_tensorcnt)
  __builtin_amdgcn_s_wait_tensorcnt(0);
#else
  asm volatile("s_wait_tensorcnt 0x0" ::: "memory");
#endif
}

DEV void tdm_load_2d(unsigned lds_addr, const void* gptr,
                     unsigned tile_w, unsigned tile_h, unsigned stride_elems) {
  unsigned long long ga = (unsigned long long)(size_t)gptr;
  // D# group 0: [1:0]=count=1 | [63:32]=lds_addr | [120:64]=global_addr | [127:126]=type=2
  v4u g0;
  g0[0] = 1u;
  g0[1] = lds_addr;
  g0[2] = (unsigned)ga;
  g0[3] = ((unsigned)(ga >> 32) & 0x01FFFFFFu) | 0x80000000u;
  // D# group 1: data_size=1 (2B); tensor_dim0=tile_w @ [79:48]; tensor_dim1=tile_h
  // @ [111:80]; tile_dim0=tile_w @ [127:112]; tile_dim1=tile_h @ [143:128];
  // tensor_dim0_stride @ [207:160].
  v8i g1;
  g1[0] = (int)(1u << 16);
  g1[1] = (int)((tile_w & 0xFFFFu) << 16);
  g1[2] = (int)((tile_h & 0xFFFFu) << 16);
  g1[3] = (int)((tile_w & 0xFFFFu) << 16);
  g1[4] = (int)(tile_h & 0xFFFFu);
  g1[5] = (int)stride_elems;
  g1[6] = 0;
  g1[7] = 0;
  v4i gz = {0, 0, 0, 0};
#if __has_builtin(__builtin_amdgcn_tensor_load_to_lds)
# if defined(__clang_major__) && __clang_major__ >= 23
  v8i gz8 = {0, 0, 0, 0, 0, 0, 0, 0};
  __builtin_amdgcn_tensor_load_to_lds(g0, g1, gz, gz, gz8, 0);
# else
  __builtin_amdgcn_tensor_load_to_lds(g0, g1, gz, gz, 0);
# endif
#else
  asm volatile("tensor_load_to_lds %0, %1, %2, %3"
               :: "s"(g0), "s"(g1), "s"(gz), "s"(gz) : "memory");
#endif
}

// ---------------------------------------------------------------------------
// C[M,N] = A[M,K] * W[N,K]^T + bias.  A is f32 (Af) or bf16 (Ab).
// Output is bf16 (Cb) or f32 (Cf).  M,N multiples of 128; K multiple of 32.
// Block: 256 threads = 8 waves; block tile 128x128; wave tile 32x64.
// ---------------------------------------------------------------------------
__global__ __launch_bounds__(256) void gemm_bias_kernel(
    const float* __restrict__ Af, const unsigned short* __restrict__ Ab,
    const float* __restrict__ W, const float* __restrict__ bias,
    float* __restrict__ Cf, unsigned short* __restrict__ Cb,
    int M, int N, int K)
{
  __shared__ unsigned short As[128 * 32];   // [m][k]
  __shared__ unsigned short Bs[128 * 32];   // [n][k]  (N-major => B-frag reads contiguous)
  const int tid = threadIdx.x, lane = tid & 31, wid = tid >> 5;
  const int m0 = blockIdx.y * 128, n0 = blockIdx.x * 128;
  const int wy = wid & 3, wx = wid >> 2;    // wave tile: rows wy*32, cols wx*64

  v8f z = {0.f, 0.f, 0.f, 0.f, 0.f, 0.f, 0.f, 0.f};
  v8f acc[2][4];
#pragma unroll
  for (int i = 0; i < 2; ++i)
#pragma unroll
    for (int j = 0; j < 4; ++j) acc[i][j] = z;

  const int row = tid >> 1, cb = (tid & 1) * 16;   // 256 threads fill 128x32
  for (int kc = 0; kc < K; kc += 32) {
    __syncthreads();
    if (Ab) {
      const uint4* src = (const uint4*)(Ab + (size_t)(m0 + row) * K + kc + cb);
      uint4* dst = (uint4*)(As + row * 32 + cb);
      dst[0] = src[0]; dst[1] = src[1];
    } else {
      const float4* src = (const float4*)(Af + (size_t)(m0 + row) * K + kc + cb);
      unsigned* dst = (unsigned*)(As + row * 32 + cb);
#pragma unroll
      for (int q = 0; q < 4; ++q) {
        float4 f = src[q];
        dst[q * 2 + 0] = pack2bf(f.x, f.y);
        dst[q * 2 + 1] = pack2bf(f.z, f.w);
      }
    }
    {
      const float4* src = (const float4*)(W + (size_t)(n0 + row) * K + kc + cb);
      unsigned* dst = (unsigned*)(Bs + row * 32 + cb);
#pragma unroll
      for (int q = 0; q < 4; ++q) {
        float4 f = src[q];
        dst[q * 2 + 0] = pack2bf(f.x, f.y);
        dst[q * 2 + 1] = pack2bf(f.z, f.w);
      }
    }
    if (kc + 32 < K) {
      if (Ab) __builtin_prefetch(Ab + (size_t)(m0 + row) * K + kc + 32 + cb, 0, 1);
      else    __builtin_prefetch(Af + (size_t)(m0 + row) * K + kc + 32 + cb, 0, 1);
      __builtin_prefetch(W + (size_t)(n0 + row) * K + kc + 32 + cb, 0, 1);
    }
    __syncthreads();

    v16bf a0 = load_a_frag(As, 32, wy * 32, 0, lane);
    v16bf a1 = load_a_frag(As, 32, wy * 32 + 16, 0, lane);
#pragma unroll
    for (int j = 0; j < 4; ++j) {
      v16bf bf = load_b_frag(Bs, 32, wx * 64 + j * 16, 0, lane);
      acc[0][j] = wmma_bf16(a0, bf, acc[0][j]);
      acc[1][j] = wmma_bf16(a1, bf, acc[1][j]);
    }
  }

  // Epilogue. C layout: VGPR r = row r (lanes 0-15) / r+8 (lanes 16-31), col = lane&15.
  const int rl_hi = (lane & 16) ? 8 : 0, nl = lane & 15;
#pragma unroll
  for (int i = 0; i < 2; ++i)
#pragma unroll
    for (int j = 0; j < 4; ++j)
#pragma unroll
      for (int r = 0; r < 8; ++r) {
        int rr = m0 + wy * 32 + i * 16 + r + rl_hi;
        int cc = n0 + wx * 64 + j * 16 + nl;
        float v = acc[i][j][r] + bias[cc];
        if (Cb) Cb[(size_t)rr * N + cc] = f2bf(v);
        else    Cf[(size_t)rr * N + cc] = v;
      }
}

// ---------------------------------------------------------------------------
// Flash attention: one block per (b, h, q-tile of 128 rows); 8 waves,
// wave w owns query rows [16w, 16w+16).  Online softmax, bf16 WMMA.
// Q and K tiles staged by the Tensor Data Mover; V staged manually (needs a
// transpose into Vt[j][t], which TDM cannot do) and overlaps the TDM copy.
// qkv layout: [B, S, 3D]; Q at col h*64, K at D + h*64, V at 2D + h*64.
// ---------------------------------------------------------------------------
__global__ __launch_bounds__(256) void attn_kernel(const unsigned short* __restrict__ qkv,
                                                   unsigned short* __restrict__ out)
{
  constexpr int S = 2048, TD = 3072, D = 1024;
  __shared__ unsigned short Qs[128 * 64];   // [m][d]
  __shared__ unsigned short Ks[128 * 64];   // [n][d]  (N-major for B-frags of QK^T)
  __shared__ unsigned short Vt[64 * 128];   // [j][t]  (N-major for B-frags of P*V)
  __shared__ unsigned short Ps[8 * 16 * 64];// per-wave P scratch (C-layout -> A-frag)
  const int tid = threadIdx.x, lane = tid & 31, wid = tid >> 5;
  const int b = blockIdx.y >> 4, h = blockIdx.y & 15;
  const int q0 = blockIdx.x * 128;
  const size_t base = (size_t)b * S * TD + (size_t)h * 64;
  const float scale = 0.125f;               // 1/sqrt(64)

  // stage Q tile via TDM (wave 0 issues; barrier releases the readers)
  if (wid == 0) {
    tdm_load_2d((unsigned)(size_t)Qs, qkv + base + (size_t)q0 * TD, 64, 128, TD);
    tdm_wait0();
  }
  __syncthreads();

  v16bf aq[2];
  aq[0] = load_a_frag(Qs, 64, wid * 16, 0, lane);
  aq[1] = load_a_frag(Qs, 64, wid * 16, 32, lane);

  v8f z = {0.f, 0.f, 0.f, 0.f, 0.f, 0.f, 0.f, 0.f};
  v8f o[4];
  float mr[8], lr[8];
#pragma unroll
  for (int j = 0; j < 4; ++j) o[j] = z;
#pragma unroll
  for (int r = 0; r < 8; ++r) { mr[r] = -3.0e38f; lr[r] = 0.f; }

  for (int kt = 0; kt < S; kt += 128) {
    __syncthreads();                        // all consumers of Ks/Vt done
    if (wid == 0)                           // async K tile -> LDS
      tdm_load_2d((unsigned)(size_t)Ks, qkv + base + (size_t)D + (size_t)kt * TD, 64, 128, TD);
    { // stage V tile transposed: Vt[j][t]  (overlaps the TDM copy)
      int row = tid >> 1, cb2 = (tid & 1) * 32;
      const unsigned short* src = qkv + base + 2 * (size_t)D + (size_t)(kt + row) * TD + cb2;
#pragma unroll
      for (int i = 0; i < 32; ++i) Vt[(cb2 + i) * 128 + row] = src[i];
    }
    if (wid == 0) tdm_wait0();
    __syncthreads();

    // S = Q K^T : 16 rows x 128 cols per wave (8 tiles x 2 K-steps)
    v8f sc[8];
#pragma unroll
    for (int j = 0; j < 8; ++j) {
      v8f s = z;
      s = wmma_bf16(aq[0], load_b_frag(Ks, 64, j * 16, 0, lane), s);
      s = wmma_bf16(aq[1], load_b_frag(Ks, 64, j * 16, 32, lane), s);
      sc[j] = s;
    }

    // online softmax: reduce across the 16 lanes of each half (wave32 shuffles)
    float alpha[8], lsum[8];
#pragma unroll
    for (int r = 0; r < 8; ++r) {
      float v = -3.0e38f;
#pragma unroll
      for (int j = 0; j < 8; ++j) v = fmaxf(v, sc[j][r]);
      v = fmaxf(v, __shfl_xor(v, 1, 32));
      v = fmaxf(v, __shfl_xor(v, 2, 32));
      v = fmaxf(v, __shfl_xor(v, 4, 32));
      v = fmaxf(v, __shfl_xor(v, 8, 32));
      float mnew = fmaxf(mr[r], v * scale);
      alpha[r] = __expf(mr[r] - mnew);
      mr[r] = mnew;
      lsum[r] = 0.f;
    }
#pragma unroll
    for (int jo = 0; jo < 4; ++jo)
#pragma unroll
      for (int e = 0; e < 8; ++e) o[jo][e] *= alpha[e];

    // P (bf16) through per-wave LDS scratch, then P*V, half (64 K-vals) at a time
    unsigned short* pw = Ps + wid * (16 * 64);
    const int rl_hi = (lane & 16) ? 8 : 0, nl = lane & 15;
#pragma unroll
    for (int half = 0; half < 2; ++half) {
#pragma unroll
      for (int j4 = 0; j4 < 4; ++j4) {
        int j = half * 4 + j4;
#pragma unroll
        for (int r = 0; r < 8; ++r) {
          float p = __expf(sc[j][r] * scale - mr[r]);
          lsum[r] += p;
          pw[(r + rl_hi) * 64 + j4 * 16 + nl] = f2bf(p);
        }
      }
#pragma unroll
      for (int k0 = 0; k0 < 2; ++k0) {
        v16bf ap = load_a_frag(pw, 64, 0, k0 * 32, lane);
#pragma unroll
        for (int jo = 0; jo < 4; ++jo) {
          v16bf bv = load_b_frag(Vt, 128, jo * 16, half * 64 + k0 * 32, lane);
          o[jo] = wmma_bf16(ap, bv, o[jo]);
        }
      }
    }
#pragma unroll
    for (int r = 0; r < 8; ++r) {
      float s = lsum[r];
      s += __shfl_xor(s, 1, 32);
      s += __shfl_xor(s, 2, 32);
      s += __shfl_xor(s, 4, 32);
      s += __shfl_xor(s, 8, 32);
      lr[r] = lr[r] * alpha[r] + s;
    }
  }

  // normalize and store O as bf16 into [B*S, D] at col h*64
  const int rl_hi = (lane & 16) ? 8 : 0, nl = lane & 15;
#pragma unroll
  for (int r = 0; r < 8; ++r) {
    float inv = 1.0f / lr[r];
    int row = q0 + wid * 16 + r + rl_hi;
    size_t ob = ((size_t)b * S + row) * D + h * 64;
#pragma unroll
    for (int jo = 0; jo < 4; ++jo)
      out[ob + jo * 16 + nl] = f2bf(o[jo][r] * inv);
  }
}

extern "C" void kernel_launch(void* const* d_in, const int* in_sizes, int n_in,
                              void* d_out, int out_size, void* d_ws, size_t ws_size,
                              hipStream_t stream) {
  (void)in_sizes; (void)n_in; (void)out_size; (void)ws_size;
  const float* x  = (const float*)d_in[0];   // [4,2048,1024]
  const float* wi = (const float*)d_in[1];   // [3072,1024]
  const float* bi = (const float*)d_in[2];   // [3072]
  const float* wo = (const float*)d_in[3];   // [1024,1024]
  const float* bo = (const float*)d_in[4];   // [1024]
  // d_in[5] = causal_mask (0) -> no masking path needed

  const int B = 4, S = 2048, D = 1024;
  unsigned short* qkv  = (unsigned short*)d_ws;             // bf16 [B*S, 3D]  (48 MB)
  unsigned short* attn = qkv + (size_t)B * S * 3 * D;       // bf16 [B*S, D]   (16 MB)

  // qkv = x @ wi^T + bi
  gemm_bias_kernel<<<dim3(3 * D / 128, B * S / 128), 256, 0, stream>>>(
      x, nullptr, wi, bi, nullptr, qkv, B * S, 3 * D, D);
  // attention
  attn_kernel<<<dim3(S / 128, B * 16), 256, 0, stream>>>(qkv, attn);
  // out = attn @ wo^T + bo
  gemm_bias_kernel<<<dim3(D / 128, B * S / 128), 256, 0, stream>>>(
      nullptr, attn, wo, bo, (float*)d_out, nullptr, B * S, D, D);
}
